// MetaBaseline_89429809037490
// MI455X (gfx1250) — matrix operand order repaired and checked by reading, not compiled, not run on hardware
//
#include <hip/hip_runtime.h>
#include <hip/hip_bf16.h>
#include <stdint.h>

#define B_    4
#define Q_    75
#define HW_   100
#define D_    640
#define WAY_  5
#define SHOT_ 5
#define NREAL_ 500   // shot*HW
#define NPAD_  512   // padded to 32 WMMA column tiles
#define TOPK_  5

typedef __attribute__((ext_vector_type(16))) __bf16 v16bf;
typedef __attribute__((ext_vector_type(8)))  float  v8f;

// ---------- helpers ----------
__device__ __forceinline__ unsigned short f2bf(float x) {
    unsigned u = __float_as_uint(x);
    u += 0x7FFFu + ((u >> 16) & 1u);          // round-to-nearest-even
    return (unsigned short)(u >> 16);
}
__device__ __forceinline__ float bf2f(unsigned short h) {
    return __uint_as_float(((unsigned)h) << 16);
}
__device__ __forceinline__ float wave_sum(float v) {
#pragma unroll
    for (int off = 16; off; off >>= 1) v += __shfl_xor(v, off, 32);
    return v;
}

__device__ __forceinline__ v8f wmma_bf(v16bf a, v16bf b, v8f c) {
    // v_wmma_f32_16x16x32_bf16  (neg_a, A, neg_b, B, c_mod, C, reuse_a, reuse_b)
    return __builtin_amdgcn_wmma_f32_16x16x32_bf16(false, a, false, b, (short)0, c,
                                                   false, false);
}

// A 16x32 bf16 fragment, contiguous k order used here (B matches): lane lo (0-15) row M=lane, K 0..15;
// lane hi (16-31) same row, K 16..31.  kd = k0/2 (dword index).
__device__ __forceinline__ v16bf load_fragA(const unsigned* rowp, int kd, int hi) {
    union { v16bf v; unsigned u[8]; } f;
    int base = kd + (hi << 3);
#pragma unroll
    for (int j = 0; j < 8; ++j) f.u[j] = rowp[base + j];
    return f.v;
}
// B matrix 32x16 bf16 fragment from a staged LDS row (64 halves = one 64-K chunk):
// lane lo holds K=0..15, lane hi K=16..31 of the 32-K step ks -> 8 contiguous dwords.
__device__ __forceinline__ v16bf load_fragB_lds(const unsigned short* rowp, int ks, int hi) {
    union { v16bf v; unsigned u[8]; } f;
    const unsigned* p = (const unsigned*)rowp;
    int base = (ks << 4) + (hi << 3);
#pragma unroll
    for (int j = 0; j < 8; ++j) f.u[j] = p[base + j];
    return f.v;
}

// Async-stage one 64-row x 64-K (8 KB) B sub-tile into LDS (whole 128-thread block).
// Each thread issues 4 x GLOBAL_LOAD_ASYNC_TO_LDS_B128 (ASYNCcnt-tracked).
__device__ __forceinline__ void stage_b(const unsigned short* __restrict__ Bbase,
                                        unsigned short (*buf)[64], int n0, int kc, int tid) {
#pragma unroll
    for (int j = 0; j < 4; ++j) {
        int s   = tid + (j << 7);          // 0..511 16-byte segments
        int r   = s >> 3;                  // staged row (support column) 0..63
        int seg = s & 7;                   // 16B segment within the 128B row
        const unsigned short* g = Bbase + (size_t)(n0 + r) * D_ + kc * 64 + seg * 8;
        unsigned lds = (unsigned)(uintptr_t)&buf[r][seg * 8];
        uint64_t ga  = (uint64_t)(uintptr_t)g;
        asm volatile("global_load_async_to_lds_b128 %0, %1, off"
                     :: "v"(lds), "v"(ga) : "memory");
    }
}

// ---------- phase 1: L2 normalize -> bf16 ----------
__global__ __launch_bounds__(256) void k_norm_query(const float* __restrict__ x,
                                                    unsigned short* __restrict__ out) {
    int wv = threadIdx.x >> 5, lane = threadIdx.x & 31;
    int row = blockIdx.x * 8 + wv;               // rows = B_*Q_*HW_ = 30000
    const float* xr = x + (size_t)row * D_;
    float ss = 0.f;
    for (int i = lane; i < D_; i += 32) { float v = xr[i]; ss += v * v; }
    ss = wave_sum(ss);
    float r = rsqrtf(ss);
    unsigned short* o = out + (size_t)row * D_;
    for (int i = lane; i < D_; i += 32) o[i] = f2bf(xr[i] * r);
}

__global__ __launch_bounds__(256) void k_norm_shot(const float* __restrict__ x,
                                                   unsigned short* __restrict__ out) {
    int wv = threadIdx.x >> 5, lane = threadIdx.x & 31;
    int row = blockIdx.x * 8 + wv;               // rows = B_*WAY_*NPAD_ = 10240
    int n = row & (NPAD_ - 1);
    int bw = row >> 9;
    unsigned short* o = out + (size_t)row * D_;
    if (n >= NREAL_) {                           // zero padding rows
        for (int i = lane; i < D_; i += 32) o[i] = 0;
        return;
    }
    const float* xr = x + ((size_t)bw * NREAL_ + n) * D_;   // [b,w,s,h,:] contiguous
    float ss = 0.f;
    for (int i = lane; i < D_; i += 32) { float v = xr[i]; ss += v * v; }
    ss = wave_sum(ss);
    float r = rsqrtf(ss);
    for (int i = lane; i < D_; i += 32) o[i] = f2bf(xr[i] * r);
}

// ---------- phase 2: means + cosine branch ----------
__global__ void k_qmean(const unsigned short* __restrict__ qn, float* __restrict__ qmean) {
    int bq = blockIdx.x;                          // B_*Q_
    const unsigned short* base = qn + (size_t)bq * HW_ * D_;
    for (int d = threadIdx.x; d < D_; d += blockDim.x) {
        float s = 0.f;
        for (int h = 0; h < HW_; ++h) s += bf2f(base[(size_t)h * D_ + d]);
        qmean[(size_t)bq * D_ + d] = s * (1.0f / HW_);
    }
}

__global__ void k_proto(const unsigned short* __restrict__ sp, float* __restrict__ proto) {
    int bw = blockIdx.x;                          // B_*WAY_
    const unsigned short* base = sp + (size_t)bw * NPAD_ * D_;
    for (int d = threadIdx.x; d < D_; d += blockDim.x) {
        float s = 0.f;
        for (int n = 0; n < NREAL_; ++n) s += bf2f(base[(size_t)n * D_ + d]);
        proto[(size_t)bw * D_ + d] = s * (1.0f / NREAL_);
    }
}

__global__ __launch_bounds__(128) void k_cos(const float* __restrict__ qmean,
                                             const float* __restrict__ proto,
                                             float* __restrict__ feat) {
    int gw = blockIdx.x * 4 + (threadIdx.x >> 5); // 1500 waves
    int lane = threadIdx.x & 31;
    int w = gw % WAY_, bq = gw / WAY_, b = bq / Q_;
    const float* qm = qmean + (size_t)bq * D_;
    const float* pr = proto + ((size_t)b * WAY_ + w) * D_;
    float s = 0.f;
    for (int i = lane; i < D_; i += 32) s += qm[i] * pr[i];
    s = wave_sum(s);
    if (lane == 0) feat[bq * (2 * WAY_) + w] = s;
}

// ---------- phase 3: fused WMMA GEMM + top-5 (DN4 branch) ----------
// One block per (b,q,w): 128 rows (4 waves x 2 M-tiles, clamped to 100) x 512 cols.
// B sub-tiles are double-buffered in LDS via async global->LDS copies so that the
// ASYNCcnt copy of k-chunk kc+1 overlaps the 16 WMMAs of k-chunk kc.
__global__ __launch_bounds__(128) void k_dn4_wmma(const unsigned short* __restrict__ qn,
                                                  const unsigned short* __restrict__ sp,
                                                  float* __restrict__ feat) {
    __shared__ unsigned short bbuf[2][64][64];  // 16 KB ping-pong B tile (64 cols x 64 K)
    __shared__ float sc[4][32][65];             // per-wave 32x64 score tile (+pad)
    __shared__ float wsums[4];

    int idx = blockIdx.x;                       // (b,q,w) = B_*Q_*WAY_ blocks
    int w  = idx % WAY_;
    int bq = idx / WAY_;
    int b  = bq / Q_;
    const unsigned short* Abase = qn + (size_t)bq * HW_ * D_;
    const unsigned short* Bbase = sp + ((size_t)b * WAY_ + w) * NPAD_ * D_;

    int tid = threadIdx.x, wv = tid >> 5, lane = tid & 31;
    int nlo = lane & 15, hi = lane >> 4;

    // this wave's two M-tiles: rows wv*32 .. wv*32+31 (clamped into [0,99])
    int r0 = wv * 32 + nlo;      if (r0 > HW_ - 1) r0 = HW_ - 1;
    int r1 = wv * 32 + 16 + nlo; if (r1 > HW_ - 1) r1 = HW_ - 1;
    const unsigned* Arow0 = (const unsigned*)(Abase + (size_t)r0 * D_);
    const unsigned* Arow1 = (const unsigned*)(Abase + (size_t)r1 * D_);

    float t0 = -1e30f, t1 = -1e30f, t2 = -1e30f, t3 = -1e30f, t4 = -1e30f;

    for (int ch = 0; ch < 8; ++ch) {            // 8 chunks of 64 support columns
        int n0 = ch << 6;
        v8f acc0[4] = {{}, {}, {}, {}};         // M-tile 0, 4 N-tiles
        v8f acc1[4] = {{}, {}, {}, {}};         // M-tile 1, 4 N-tiles

        stage_b(Bbase, bbuf[0], n0, 0, tid);    // prefetch first k-chunk

        for (int kc = 0; kc < 10; ++kc) {       // 10 k-chunks of 64 (20 WMMA k-steps)
            int cur = kc & 1;
            if (kc < 9) {
                stage_b(Bbase, bbuf[1 - cur], n0, kc + 1, tid); // async prefetch next
                asm volatile("s_wait_asynccnt 4" ::: "memory"); // current tile landed
            } else {
                asm volatile("s_wait_asynccnt 0" ::: "memory");
            }
            __syncthreads();                    // tile visible to all waves
#pragma unroll
            for (int ks = 0; ks < 2; ++ks) {    // two 32-K WMMA steps per chunk
                int kd = (kc * 64 + ks * 32) >> 1;
                v16bf a0 = load_fragA(Arow0, kd, hi);
                v16bf a1 = load_fragA(Arow1, kd, hi);
#pragma unroll
                for (int nt = 0; nt < 4; ++nt) {
                    v16bf bf = load_fragB_lds(&bbuf[cur][nt * 16 + nlo][0], ks, hi);
                    acc0[nt] = wmma_bf(a0, bf, acc0[nt]);
                    acc1[nt] = wmma_bf(a1, bf, acc1[nt]);
                }
            }
            __syncthreads();                    // done reading before tile reuse
        }

        // dump both 16x64 score tiles to this wave's LDS region
        int mrow = hi << 3;                     // C layout: M = v + 8*hi, N = lane&15
#pragma unroll
        for (int v = 0; v < 8; ++v) {
#pragma unroll
            for (int nt = 0; nt < 4; ++nt) {
                sc[wv][mrow + v][nt * 16 + nlo]      = acc0[nt][v];
                sc[wv][16 + mrow + v][nt * 16 + nlo] = acc1[nt][v];
            }
        }
        asm volatile("s_wait_dscnt 0" ::: "memory"); // same-wave LDS RAW fence

        // all 32 lanes scan one row each (both M-tiles), running top-5 in registers
        int ncols = NREAL_ - n0; if (ncols > 64) ncols = 64;
        const float* rs = &sc[wv][lane][0];
        for (int c = 0; c < ncols; ++c) {
            float v = rs[c];
            if (v > t4) {
                if      (v > t0) { t4 = t3; t3 = t2; t2 = t1; t1 = t0; t0 = v; }
                else if (v > t1) { t4 = t3; t3 = t2; t2 = t1; t1 = v; }
                else if (v > t2) { t4 = t3; t3 = t2; t2 = v; }
                else if (v > t3) { t4 = t3; t3 = v; }
                else             { t4 = v; }
            }
        }
    }

    float wsum = (wv * 32 + lane < HW_) ? (t0 + t1 + t2 + t3 + t4) : 0.f;
    wsum = wave_sum(wsum);
    if (lane == 0) wsums[wv] = wsum;
    __syncthreads();
    if (tid == 0) {
        float s = wsums[0] + wsums[1] + wsums[2] + wsums[3];
        feat[bq * (2 * WAY_) + WAY_ + w] = s * (1.0f / (HW_ * TOPK_));
    }
}

// ---------- phase 4: BatchNorm (batch stats over q) + dilated conv ----------
__global__ void k_bnconv(const float* __restrict__ feat, const float* __restrict__ gamma,
                         const float* __restrict__ beta, const float* __restrict__ convw,
                         float* __restrict__ out) {
    __shared__ float bnb[B_][Q_][2 * WAY_];
    int tid = threadIdx.x;
    for (int j = tid; j < B_ * 2 * WAY_; j += blockDim.x) {
        int b = j / (2 * WAY_), f = j % (2 * WAY_);
        float m = 0.f, m2 = 0.f;
        for (int q = 0; q < Q_; ++q) {
            float v = feat[((size_t)b * Q_ + q) * (2 * WAY_) + f];
            m += v; m2 += v * v;
        }
        m *= (1.0f / Q_); m2 *= (1.0f / Q_);
        float var = m2 - m * m;                    // biased, as torch BN training
        float scl = rsqrtf(var + 1e-5f) * gamma[f];
        float bet = beta[f];
        for (int q = 0; q < Q_; ++q)
            bnb[b][q][f] = (feat[((size_t)b * Q_ + q) * (2 * WAY_) + f] - m) * scl + bet;
    }
    __syncthreads();
    float w0 = convw[0], w1 = convw[1];
    for (int j = tid; j < B_ * Q_ * WAY_; j += blockDim.x) {
        int w = j % WAY_; int bq = j / WAY_;
        int b = bq / Q_, q = bq % Q_;
        out[j] = w0 * bnb[b][q][w] + w1 * bnb[b][q][w + WAY_];
    }
}

// ---------- driver ----------
extern "C" void kernel_launch(void* const* d_in, const int* in_sizes, int n_in,
                              void* d_out, int out_size, void* d_ws, size_t ws_size,
                              hipStream_t stream) {
    (void)in_sizes; (void)n_in; (void)out_size; (void)ws_size;
    const float* xq    = (const float*)d_in[0];
    const float* xs    = (const float*)d_in[1];
    const float* gamma = (const float*)d_in[2];
    const float* beta  = (const float*)d_in[3];
    const float* convw = (const float*)d_in[4];
    float* out = (float*)d_out;

    char* ws = (char*)d_ws;
    size_t off = 0;
    unsigned short* qn = (unsigned short*)(ws + off); off += (size_t)B_ * Q_ * HW_ * D_ * 2;
    unsigned short* sp = (unsigned short*)(ws + off); off += (size_t)B_ * WAY_ * NPAD_ * D_ * 2;
    float* qmean = (float*)(ws + off); off += (size_t)B_ * Q_ * D_ * 4;
    float* proto = (float*)(ws + off); off += (size_t)B_ * WAY_ * D_ * 4;
    float* feat  = (float*)(ws + off); off += (size_t)B_ * Q_ * 2 * WAY_ * 4;

    k_norm_query<<<(B_ * Q_ * HW_) / 8,     256, 0, stream>>>(xq, qn);
    k_norm_shot <<<(B_ * WAY_ * NPAD_) / 8, 256, 0, stream>>>(xs, sp);
    k_qmean<<<B_ * Q_,   128, 0, stream>>>(qn, qmean);
    k_proto<<<B_ * WAY_, 128, 0, stream>>>(sp, proto);
    k_cos  <<<(B_ * Q_ * WAY_) / 4, 128, 0, stream>>>(qmean, proto, feat);
    k_dn4_wmma<<<B_ * Q_ * WAY_, 128, 0, stream>>>(qn, sp, feat);
    k_bnconv<<<1, 64, 0, stream>>>(feat, gamma, beta, convw, out);
}
